// MotifPool_77137612636204
// MI455X (gfx1250) — matrix operand order repaired and checked by reading, not compiled, not run on hardware
//
#include <hip/hip_runtime.h>
#include <hip/hip_bf16.h>

typedef __attribute__((ext_vector_type(16))) __bf16 v16bf;
typedef __attribute__((ext_vector_type(8)))  __bf16 v8bf;
typedef __attribute__((ext_vector_type(8)))  float  v8f;
typedef __attribute__((ext_vector_type(4)))  float  v4f;

#define HID 512
#define NHEAD 8
#define HC 64
#define MDIM 128
#define BDIM 128
#define SLOTS (BDIM * MDIM)
#define LN_EPS 1e-5f

// Build a 16-element bf16 fragment from two contiguous 8-element (16B) chunks.
__device__ __forceinline__ v16bf frag16(const __bf16* p0, const __bf16* p1) {
    v8bf lo = *(const v8bf*)p0;
    v8bf hi = *(const v8bf*)p1;
    return __builtin_shufflevector(lo, hi, 0, 1, 2, 3, 4, 5, 6, 7,
                                   8, 9, 10, 11, 12, 13, 14, 15);
}

// ---------------- block reductions ----------------
__device__ __forceinline__ float breduce_sum(float v, float* red) {
    int t = threadIdx.x;
    red[t] = v; __syncthreads();
    for (int s = blockDim.x >> 1; s > 0; s >>= 1) {
        if (t < s) red[t] += red[t + s];
        __syncthreads();
    }
    float r = red[0]; __syncthreads();
    return r;
}
__device__ __forceinline__ float breduce_max(float v, float* red) {
    int t = threadIdx.x;
    red[t] = v; __syncthreads();
    for (int s = blockDim.x >> 1; s > 0; s >>= 1) {
        if (t < s) red[t] = fmaxf(red[t], red[t + s]);
        __syncthreads();
    }
    float r = red[0]; __syncthreads();
    return r;
}

// ---------------- f32 -> bf16 convert, 8 elems/thread (total % 8 == 0) ----------------
__global__ void conv_bf16_kernel(const float* __restrict__ src,
                                 __bf16* __restrict__ dst, int total8) {
    for (int i = blockIdx.x * blockDim.x + threadIdx.x; i < total8;
         i += gridDim.x * blockDim.x) {
        const v4f* s4 = (const v4f*)(src + i * 8);
        v4f a = s4[0], b = s4[1];
        v8bf o;
        o[0] = (__bf16)a.x; o[1] = (__bf16)a.y; o[2] = (__bf16)a.z; o[3] = (__bf16)a.w;
        o[4] = (__bf16)b.x; o[5] = (__bf16)b.y; o[6] = (__bf16)b.z; o[7] = (__bf16)b.w;
        *(v8bf*)(dst + i * 8) = o;
    }
}

// ---------------- init: cx0 = clique_x ; sums = 0 ; cnt = 0 (vec4) ----------------
__global__ void init_kernel(const float* __restrict__ clique_x, float* __restrict__ cx0,
                            float* __restrict__ sums, float* __restrict__ cnt, int total4) {
    v4f z = {0.f, 0.f, 0.f, 0.f};
    for (int i = blockIdx.x * blockDim.x + threadIdx.x; i < total4;
         i += gridDim.x * blockDim.x) {
        ((v4f*)cx0)[i] = ((const v4f*)clique_x)[i];
        ((v4f*)sums)[i] = z;
        if (i < SLOTS / 4) ((v4f*)cnt)[i] = z;
    }
}

// ---------------- PE encode + LN + scatter-add into cx0 ----------------
__global__ void pe_kernel(const float* __restrict__ clique_pe,
                          const float* __restrict__ pe_lin_w,
                          const float* __restrict__ pw, const float* __restrict__ pb,
                          const int* __restrict__ flat_index,
                          float* __restrict__ cx0, int Nc) {
    __shared__ float red[256];
    int i = blockIdx.x;
    if (i >= Nc) return;
    int t = threadIdx.x;
    const float* pe = clique_pe + i * 20;
    float y0 = 0.f, y1 = 0.f;
#pragma unroll
    for (int p = 0; p < 20; ++p) {
        float pv = pe[p];
        y0 += pv * pe_lin_w[t * 20 + p];
        y1 += pv * pe_lin_w[(t + 256) * 20 + p];
    }
    float mean = breduce_sum(y0 + y1, red) * (1.0f / HID);
    float d0 = y0 - mean, d1 = y1 - mean;
    float var = breduce_sum(d0 * d0 + d1 * d1, red) * (1.0f / HID);
    float inv = rsqrtf(var + LN_EPS);
    int fi = flat_index[i];
    cx0[fi * HID + t]       += d0 * inv * pw[t] + pb[t];
    cx0[fi * HID + t + 256] += d1 * inv * pw[t + 256] + pb[t + 256];
}

// ---------------- atom -> clique scatter add (native f32 atomics) ----------------
__global__ void edge_kernel(const float* __restrict__ x, const int* __restrict__ row,
                            const int* __restrict__ flat, float* __restrict__ sums,
                            float* __restrict__ cnt, int E) {
    int e = blockIdx.x;
    if (e >= E) return;
    int r = row[e], f = flat[e];
    const float* xp = x + (size_t)r * HID;
    float* sp = sums + (size_t)f * HID;
    for (int j = threadIdx.x; j < HID; j += blockDim.x)
        atomicAdd(&sp[j], xp[j]);
    if (threadIdx.x == 0) atomicAdd(&cnt[f], 1.0f);
}

// ---------------- mean -> bf16, 8 elems/thread ----------------
__global__ void mean_kernel(const float* __restrict__ sums, const float* __restrict__ cnt,
                            __bf16* __restrict__ meanA, int total8) {
    for (int i = blockIdx.x * blockDim.x + threadIdx.x; i < total8;
         i += gridDim.x * blockDim.x) {
        int base = i * 8;
        float rcp = 1.0f / fmaxf(cnt[base >> 9], 1.0f);   // 8 elems share a row (512 % 8 == 0)
        const v4f* s4 = (const v4f*)(sums + base);
        v4f a = s4[0], b = s4[1];
        v8bf o;
        o[0] = (__bf16)(a.x * rcp); o[1] = (__bf16)(a.y * rcp);
        o[2] = (__bf16)(a.z * rcp); o[3] = (__bf16)(a.w * rcp);
        o[4] = (__bf16)(b.x * rcp); o[5] = (__bf16)(b.y * rcp);
        o[6] = (__bf16)(b.z * rcp); o[7] = (__bf16)(b.w * rcp);
        *(v8bf*)(meanA + base) = o;
    }
}

// ---------------- WMMA bf16 GEMM: C[M,N] = A[M,K] @ W[N,K]^T (+bias) ----------------
// A: bf16 [Mt,K] lda ; W: bf16 [N,K] row-major ldk (original torch layout).
// Per-lane fragments are contiguous 16B chunks (ISA 7.12.2):
//   A: K = hi*8 + 0..7 and 16 + hi*8 + 0..7 at row m0+lane%16
//   B: K = hi*8 + 0..7 and 16 + hi*8 + 0..7 at W-row n0+lane%16  (W^T column = W row)
// MODE 0: f32 store ; 1: bf16 store ; 2: relu -> bf16 store
template <int MODE>
__global__ __launch_bounds__(256) void gemm_bf16_wmma(
        const __bf16* __restrict__ A, int lda,
        const __bf16* __restrict__ W, int ldk,
        float* __restrict__ Cf, __bf16* __restrict__ Cb, int ldc,
        int Ntot, int K, const float* __restrict__ bias, int tiles_total) {
    int wave = threadIdx.x >> 5;
    int lane = threadIdx.x & 31;
    int tile = blockIdx.x * (blockDim.x >> 5) + wave;
    if (tile >= tiles_total) return;          // wave-uniform: EXEC stays all-ones
    int ntn = Ntot >> 4;
    int m0 = (tile / ntn) << 4;
    int n0 = (tile % ntn) << 4;
    int mr = lane & 15, hi = lane >> 4;

    const __bf16* Arow = A + (m0 + mr) * lda + hi * 8;
    const __bf16* Wrow = W + (n0 + mr) * ldk + hi * 8;

    v8f acc = {};
    for (int k0 = 0; k0 < K; k0 += 32) {
        __builtin_prefetch(Wrow + k0 + 128, 0, 1);
        v16bf a = frag16(Arow + k0, Arow + k0 + 16);
        v16bf b = frag16(Wrow + k0, Wrow + k0 + 16);
        acc = __builtin_amdgcn_wmma_f32_16x16x32_bf16(false, a, false, b,
                                                      (short)0, acc, false, false);
    }
    float bv = bias ? bias[n0 + mr] : 0.0f;
    int nc = n0 + mr;
#pragma unroll
    for (int v = 0; v < 8; ++v) {
        int cidx = (m0 + v + hi * 8) * ldc + nc;
        float val = acc[v] + bv;
        if (MODE == 0)      Cf[cidx] = val;
        else if (MODE == 1) Cb[cidx] = (__bf16)val;
        else                Cb[cidx] = (__bf16)fmaxf(val, 0.0f);
    }
}

// ---------------- LN(hx) ; cx0 += ; emit bf16 copy ----------------
__global__ void ln_add_kernel(const float* __restrict__ hx, float* __restrict__ cx0,
                              __bf16* __restrict__ cx0b,
                              const float* __restrict__ w, const float* __restrict__ b) {
    __shared__ float red[256];
    int rowi = blockIdx.x, t = threadIdx.x;
    float v0 = hx[rowi * HID + t], v1 = hx[rowi * HID + t + 256];
    float mean = breduce_sum(v0 + v1, red) * (1.0f / HID);
    float d0 = v0 - mean, d1 = v1 - mean;
    float var = breduce_sum(d0 * d0 + d1 * d1, red) * (1.0f / HID);
    float inv = rsqrtf(var + LN_EPS);
    float c0 = cx0[rowi * HID + t]       + d0 * inv * w[t] + b[t];
    float c1 = cx0[rowi * HID + t + 256] + d1 * inv * w[t + 256] + b[t + 256];
    cx0[rowi * HID + t] = c0;          cx0b[rowi * HID + t] = (__bf16)c0;
    cx0[rowi * HID + t + 256] = c1;    cx0b[rowi * HID + t + 256] = (__bf16)c1;
}

// ---------------- attention: one workgroup per (b,h) ----------------
// LDS: qs[32][64], ksm[128][64] (K as [n][c] -> B-frag contiguous),
//      vts[64][128] (V transposed -> B-frag contiguous),
//      sc[32][128] f32 scores, at[32][128] bf16 attn weights.  Total 60 KB.
__global__ __launch_bounds__(256) void attn_kernel(const __bf16* __restrict__ qkvb,
                                                   __bf16* __restrict__ ctxb,
                                                   const unsigned char* __restrict__ mask) {
    __shared__ __bf16 qs[32 * HC];        //  4 KB
    __shared__ __bf16 ksm[MDIM * HC];     // 16 KB
    __shared__ __bf16 vts[HC * MDIM];     // 16 KB  (vts[c][k] = V[k][c])
    __shared__ float  sc[32 * MDIM];      // 16 KB
    __shared__ __bf16 at[32 * MDIM];      //  8 KB
    int b = blockIdx.x >> 3, h = blockIdx.x & 7;
    int tid = threadIdx.x;
    int lane = tid & 31, wave = tid >> 5;
    int mr = lane & 15, hi = lane >> 4;

    // stage K and V^T : 128-bit global loads (contiguous bf16 within a qkv row)
    for (int idx = tid; idx < MDIM * HC / 8; idx += 256) {
        int mrow = idx >> 3;
        int c8 = (idx & 7) * 8;
        const __bf16* rowp = qkvb + (b * MDIM + mrow) * (3 * HID);
        v8bf kv = *(const v8bf*)(rowp + HID + h * HC + c8);
        *(v8bf*)(ksm + mrow * HC + c8) = kv;
        v8bf vv = *(const v8bf*)(rowp + 2 * HID + h * HC + c8);
#pragma unroll
        for (int j = 0; j < 8; ++j) vts[(c8 + j) * MDIM + mrow] = vv[j];
    }

    for (int chunk = 0; chunk < 4; ++chunk) {
        int q0 = chunk * 32;
        __syncthreads();                  // K/V loaded; prev chunk done reading at/sc
        {                                 // stage Q chunk (one v8bf per thread)
            int mrow = tid >> 3;
            int c8 = (tid & 7) * 8;
            v8bf qv = *(const v8bf*)(qkvb + (b * MDIM + q0 + mrow) * (3 * HID) + h * HC + c8);
            *(v8bf*)(qs + mrow * HC + c8) = qv;
        }
        __syncthreads();

        // scores = q k^T / 8 , masked ; 2x8 tiles -> 2 per wave
        for (int i = 0; i < 2; ++i) {
            int tileid = wave * 2 + i;
            int mt = tileid >> 3, nt = tileid & 7;
            int m0 = mt * 16, n0 = nt * 16;
            const __bf16* Arow = qs + (m0 + mr) * HC + hi * 8;
            const __bf16* Brow = ksm + (n0 + mr) * HC + hi * 8;
            v8f acc = {};
            for (int k0 = 0; k0 < HC; k0 += 32) {
                v16bf a = frag16(Arow + k0, Arow + k0 + 16);
                v16bf bb = frag16(Brow + k0, Brow + k0 + 16);
                acc = __builtin_amdgcn_wmma_f32_16x16x32_bf16(false, a, false, bb,
                                                              (short)0, acc, false, false);
            }
            int nc = n0 + mr;
            float mval = mask[b * MDIM + nc] ? 0.0f : -1e9f;
#pragma unroll
            for (int v = 0; v < 8; ++v)
                sc[(m0 + v + hi * 8) * MDIM + nc] = acc[v] * 0.125f + mval;
        }
        __syncthreads();

        // row softmax (f32, 32 rows) -> bf16 attn weights
        if (tid < 32) {
            float* rw = sc + tid * MDIM;
            float mx = -3.4e38f;
            for (int k = 0; k < MDIM; ++k) mx = fmaxf(mx, rw[k]);
            float s = 0.f;
            for (int k = 0; k < MDIM; ++k) { float e = __expf(rw[k] - mx); rw[k] = e; s += e; }
            float invs = 1.0f / s;
            for (int k = 0; k < MDIM; ++k) at[tid * MDIM + k] = (__bf16)(rw[k] * invs);
        }
        __syncthreads();

        // ctx = attn @ v ; 2x4 tiles -> 1 per wave
        {
            int mt = wave >> 2, nt = wave & 3;
            int m0 = mt * 16, n0 = nt * 16;
            const __bf16* Arow = at + (m0 + mr) * MDIM + hi * 8;
            const __bf16* Brow = vts + (n0 + mr) * MDIM + hi * 8;
            v8f acc = {};
            for (int k0 = 0; k0 < MDIM; k0 += 32) {
                v16bf a = frag16(Arow + k0, Arow + k0 + 16);
                v16bf bb = frag16(Brow + k0, Brow + k0 + 16);
                acc = __builtin_amdgcn_wmma_f32_16x16x32_bf16(false, a, false, bb,
                                                              (short)0, acc, false, false);
            }
            int nc = n0 + mr;
#pragma unroll
            for (int v = 0; v < 8; ++v)
                ctxb[(b * MDIM + q0 + m0 + v + hi * 8) * HID + h * HC + nc] = (__bf16)acc[v];
        }
    }
}

// ---------------- residual + mask + LN(clique_norm) -> cxn (f32 + bf16) ----------------
__global__ void ln2_kernel(const float* __restrict__ hbuf, const float* __restrict__ cx0,
                           const unsigned char* __restrict__ mask,
                           float* __restrict__ cxn, __bf16* __restrict__ cxnb,
                           const float* __restrict__ w, const float* __restrict__ b) {
    __shared__ float red[256];
    int rowi = blockIdx.x, t = threadIdx.x;
    float mk = mask[rowi] ? 1.0f : 0.0f;
    float v0 = cx0[rowi * HID + t]       + hbuf[rowi * HID + t] * mk;
    float v1 = cx0[rowi * HID + t + 256] + hbuf[rowi * HID + t + 256] * mk;
    float mean = breduce_sum(v0 + v1, red) * (1.0f / HID);
    float d0 = v0 - mean, d1 = v1 - mean;
    float var = breduce_sum(d0 * d0 + d1 * d1, red) * (1.0f / HID);
    float inv = rsqrtf(var + LN_EPS);
    float o0 = d0 * inv * w[t] + b[t];
    float o1 = d1 * inv * w[t + 256] + b[t + 256];
    cxn[rowi * HID + t] = o0;          cxnb[rowi * HID + t] = (__bf16)o0;
    cxn[rowi * HID + t + 256] = o1;    cxnb[rowi * HID + t + 256] = (__bf16)o1;
}

// ---------------- per-(b,h) score dot + masked softmax pooling ----------------
__global__ void pool_kernel(const __bf16* __restrict__ hidr, const float* __restrict__ w2,
                            const float* __restrict__ b2,
                            const unsigned char* __restrict__ mask,
                            const float* __restrict__ cxn, float* __restrict__ out) {
    __shared__ float red[MDIM];
    __shared__ float wsh[MDIM];
    int b = blockIdx.x >> 3, h = blockIdx.x & 7;
    int m = threadIdx.x;     // 128 threads
    float s = b2[h];
    const v8bf* hp = (const v8bf*)(hidr + ((size_t)h * SLOTS + b * MDIM + m) * 128);
    const float* w2h = w2 + h * 128;
    for (int d8 = 0; d8 < 16; ++d8) {
        v8bf hv = hp[d8];
#pragma unroll
        for (int j = 0; j < 8; ++j) s += (float)hv[j] * w2h[d8 * 8 + j];
    }
    if (!mask[b * MDIM + m]) s = -1e9f;
    float mx = breduce_max(s, red);
    float e = __expf(s - mx);
    float tot = breduce_sum(e, red);
    wsh[m] = e / tot;
    __syncthreads();
    if (m < HC) {
        float acc = 0.f;
        for (int k = 0; k < MDIM; ++k)
            acc += wsh[k] * cxn[(b * MDIM + k) * HID + h * HC + m];
        out[b * HID + h * HC + m] = acc;
    }
}

// =====================================================================
extern "C" void kernel_launch(void* const* d_in, const int* in_sizes, int n_in,
                              void* d_out, int out_size, void* d_ws, size_t ws_size,
                              hipStream_t stream) {
    const float* x           = (const float*)d_in[0];
    const float* clique_x    = (const float*)d_in[1];
    const float* clique_pe   = (const float*)d_in[2];
    const float* pe_lin_w    = (const float*)d_in[3];
    const float* pe_norm_w   = (const float*)d_in[4];
    const float* pe_norm_b   = (const float*)d_in[5];
    const float* atom_proj_w = (const float*)d_in[6];
    const float* atom_norm_w = (const float*)d_in[7];
    const float* atom_norm_b = (const float*)d_in[8];
    const float* in_proj_w   = (const float*)d_in[9];
    const float* in_proj_b   = (const float*)d_in[10];
    const float* out_proj_w  = (const float*)d_in[11];
    const float* out_proj_b  = (const float*)d_in[12];
    const float* cl_norm_w   = (const float*)d_in[13];
    const float* cl_norm_b   = (const float*)d_in[14];
    const float* score_w1    = (const float*)d_in[15];
    const float* score_b1    = (const float*)d_in[16];
    const float* score_w2    = (const float*)d_in[17];
    const float* score_b2    = (const float*)d_in[18];
    const int*   a2c_row     = (const int*)d_in[19];
    const int*   a2c_flat    = (const int*)d_in[20];
    const int*   flat_index  = (const int*)d_in[21];
    const unsigned char* mask = (const unsigned char*)d_in[22];
    float* out = (float*)d_out;

    const int E  = in_sizes[19];
    const int Nc = in_sizes[21];
    const int total = SLOTS * HID;

    // workspace carve-out (256B aligned)
    char* ws = (char*)d_ws;
    size_t off = 0;
    auto carve = [&](size_t bytes) {
        size_t o = off;
        off = (off + bytes + 255) & ~(size_t)255;
        return o;
    };
    float*  cx0   = (float*)(ws + carve((size_t)total * 4));
    float*  sums  = (float*)(ws + carve((size_t)total * 4));  // reused: hxbuf, hbuf
    float*  cnt   = (float*)(ws + carve((size_t)SLOTS * 4));
    __bf16* meanA = (__bf16*)(ws + carve((size_t)total * 2)); // reused: cxnb
    __bf16* cx0b  = (__bf16*)(ws + carve((size_t)total * 2));
    __bf16* qkvb  = (__bf16*)(ws + carve((size_t)SLOTS * 3 * HID * 2));
    __bf16* ctxb  = (__bf16*)(ws + carve((size_t)total * 2));
    float*  cxn   = (float*)(ws + carve((size_t)total * 4));
    __bf16* hidr  = (__bf16*)(ws + carve((size_t)NHEAD * SLOTS * 128 * 2));
    __bf16* atomWb = (__bf16*)(ws + carve((size_t)HID * HID * 2));
    __bf16* inWb   = (__bf16*)(ws + carve((size_t)HID * 3 * HID * 2));
    __bf16* outWb  = (__bf16*)(ws + carve((size_t)HID * HID * 2));
    __bf16* w1b    = (__bf16*)(ws + carve((size_t)NHEAD * 128 * HC * 2));

    float*  hxbuf = sums;   // after meanA is extracted
    float*  hbuf  = sums;   // after hx consumed
    __bf16* cxnb  = meanA;  // after GEMM(meanA) consumed

    // ---- weights -> bf16 (layout preserved: [N,K] rows = W^T columns) ----
    conv_bf16_kernel<<<(HID * HID / 8 + 255) / 256, 256, 0, stream>>>(atom_proj_w, atomWb, HID * HID / 8);
    conv_bf16_kernel<<<(3 * HID * HID / 8 + 255) / 256, 256, 0, stream>>>(in_proj_w, inWb, 3 * HID * HID / 8);
    conv_bf16_kernel<<<(HID * HID / 8 + 255) / 256, 256, 0, stream>>>(out_proj_w, outWb, HID * HID / 8);
    conv_bf16_kernel<<<(NHEAD * 128 * HC / 8 + 255) / 256, 256, 0, stream>>>(score_w1, w1b, NHEAD * 128 * HC / 8);

    // ---- init ----
    init_kernel<<<total / 4 / 256, 256, 0, stream>>>(clique_x, cx0, sums, cnt, total / 4);

    // ---- PE encode + scatter ----
    pe_kernel<<<Nc, 256, 0, stream>>>(clique_pe, pe_lin_w, pe_norm_w, pe_norm_b,
                                      flat_index, cx0, Nc);

    // ---- atom -> clique scatter mean ----
    edge_kernel<<<E, 128, 0, stream>>>(x, a2c_row, a2c_flat, sums, cnt, E);
    mean_kernel<<<total / 8 / 256, 256, 0, stream>>>(sums, cnt, meanA, total / 8);

    // ---- hx = mean @ atom_proj^T ----
    {
        int tiles = (SLOTS / 16) * (HID / 16);
        gemm_bf16_wmma<0><<<(tiles + 7) / 8, 256, 0, stream>>>(
            meanA, HID, atomWb, HID, hxbuf, nullptr, HID, HID, HID, nullptr, tiles);
    }
    ln_add_kernel<<<SLOTS, 256, 0, stream>>>(hxbuf, cx0, cx0b, atom_norm_w, atom_norm_b);

    // ---- qkv = cx0 @ in_proj^T + b ----
    {
        int tiles = (SLOTS / 16) * (3 * HID / 16);
        gemm_bf16_wmma<1><<<(tiles + 7) / 8, 256, 0, stream>>>(
            cx0b, HID, inWb, HID, nullptr, qkvb, 3 * HID, 3 * HID, HID, in_proj_b, tiles);
    }

    // ---- attention ----
    attn_kernel<<<BDIM * NHEAD, 256, 0, stream>>>(qkvb, ctxb, mask);

    // ---- out proj + residual + LN ----
    {
        int tiles = (SLOTS / 16) * (HID / 16);
        gemm_bf16_wmma<0><<<(tiles + 7) / 8, 256, 0, stream>>>(
            ctxb, HID, outWb, HID, hbuf, nullptr, HID, HID, HID, out_proj_b, tiles);
    }
    ln2_kernel<<<SLOTS, 256, 0, stream>>>(hbuf, cx0, mask, cxn, cxnb,
                                          cl_norm_w, cl_norm_b);

    // ---- per-head score MLP layer 1 (ReLU) ----
    for (int h = 0; h < NHEAD; ++h) {
        int tiles = (SLOTS / 16) * (128 / 16);
        gemm_bf16_wmma<2><<<(tiles + 7) / 8, 256, 0, stream>>>(
            cxnb + h * HC, HID, w1b + h * 128 * HC, HC,
            nullptr, hidr + (size_t)h * SLOTS * 128, 128,
            128, HC, score_b1 + h * 128, tiles);
    }

    // ---- score dot + masked softmax pooling ----
    pool_kernel<<<BDIM * NHEAD, 128, 0, stream>>>(hidr, score_w2, score_b2,
                                                  mask, cxn, out);
    (void)n_in; (void)out_size; (void)ws_size;
}
